// EncoderLayer_70866960384275
// MI455X (gfx1250) — matrix operand order repaired and checked by reading, not compiled, not run on hardware
//
#include <hip/hip_runtime.h>
#include <hip/hip_bf16.h>

#define DEV __device__ __forceinline__

typedef __attribute__((ext_vector_type(16))) __bf16          v16bf;
typedef __attribute__((ext_vector_type(8)))  float           v8f;
typedef __attribute__((ext_vector_type(8)))  unsigned short  v8u16;
typedef __attribute__((ext_vector_type(16))) unsigned short  v16u16;

constexpr int Bsz = 4, Sq = 1024, Dm = 1024, NH = 16, HD = 64, FF = 4096;
constexpr int Mrows = Bsz * Sq; // 4096

// ---------------- CDNA5 Tensor Data Mover support (probe-guarded) ----------------
#if __has_builtin(__builtin_amdgcn_tensor_load_to_lds) && __has_builtin(__builtin_amdgcn_s_wait_tensorcnt)
#define HAVE_TDM 1
#else
#define HAVE_TDM 0
#endif

#if HAVE_TDM
typedef __attribute__((ext_vector_type(4))) unsigned int v4u32;
typedef __attribute__((ext_vector_type(8))) int          v8i32;
typedef __attribute__((ext_vector_type(4))) int          v4i32;

DEV unsigned lds_off(const void* p) {
    // generic pointer to LDS: low 32 bits carry the LDS byte offset
    return (unsigned)(unsigned long long)p;
}

// 2-D tile load: tile0 contiguous elems (bf16) x tile1 rows, source row stride
// srcStrideElems; LDS padding: insert 4 DWORDs (8 elems) every 2^(padCode+1) DWORDs.
DEV void tdm_load_2d(unsigned ldsByteOff, const unsigned short* g,
                     unsigned tile0, unsigned tile1,
                     unsigned srcStrideElems, unsigned padCode) {
    unsigned long long ga = (unsigned long long)(const void*)g;
    v4u32 g0;
    g0[0] = 1u;                                                  // count=1 (valid descriptor)
    g0[1] = ldsByteOff;                                          // lds_addr
    g0[2] = (unsigned)ga;                                        // global_addr[31:0]
    g0[3] = ((unsigned)(ga >> 32) & 0x01FFFFFFu) | (2u << 30);   // global_addr[56:32] | type=2
    unsigned td0 = srcStrideElems;                               // tensor_dim0
    unsigned td1 = tile1;                                        // tensor_dim1
    v8i32 g1;
    g1[0] = (int)((1u << 16) | (1u << 20) | (padCode << 22) | (3u << 25)); // 2B data, pad_en, interval, amount=4DW
    g1[1] = (int)((td0 & 0xFFFFu) << 16);                        // tensor_dim0[15:0] @ bit48
    g1[2] = (int)(((td0 >> 16) & 0xFFFFu) | ((td1 & 0xFFFFu) << 16));
    g1[3] = (int)(((td1 >> 16) & 0xFFFFu) | ((tile0 & 0xFFFFu) << 16));   // tile_dim0
    g1[4] = (int)(tile1 & 0xFFFFu);                              // tile_dim1
    g1[5] = (int)srcStrideElems;                                 // tensor_dim0_stride[31:0]
    g1[6] = 0;
    g1[7] = 0;
    v4i32 z4 = {0, 0, 0, 0};
#if defined(__clang_major__) && (__clang_major__ >= 23)
    v8i32 z8 = {0, 0, 0, 0, 0, 0, 0, 0};
    __builtin_amdgcn_tensor_load_to_lds(g0, g1, z4, z4, z8, 0);
#else
    __builtin_amdgcn_tensor_load_to_lds(g0, g1, z4, z4, 0);
#endif
}

DEV void tdm_wait() { __builtin_amdgcn_s_wait_tensorcnt(0); }
#endif // HAVE_TDM

DEV unsigned short f2bf(float f) {
    unsigned u = __float_as_uint(f);
    u += 0x7FFFu + ((u >> 16) & 1u);
    return (unsigned short)(u >> 16);
}

DEV v16bf mkfrag(const unsigned short* p_lo, const unsigned short* p_hi) {
    v8u16 lo = *(const v8u16*)p_lo;
    v8u16 hi = *(const v8u16*)p_hi;
    v16u16 c = __builtin_shufflevector(lo, hi, 0,1,2,3,4,5,6,7,8,9,10,11,12,13,14,15);
    union { v16u16 u; v16bf b; } x;
    x.u = c;
    return x.b;
}

DEV v8f wmma_bf16(v16bf a, v16bf b, v8f c) {
    return __builtin_amdgcn_wmma_f32_16x16x32_bf16(false, a, false, b, (short)0, c, false, false);
}

DEV v8f vzero8() {
    v8f z;
    #pragma unroll
    for (int i = 0; i < 8; ++i) z[i] = 0.f;
    return z;
}

// ---------------- conversion kernels ----------------

__global__ __launch_bounds__(256) void k_f32_to_bf16(const float* __restrict__ in,
                                                     unsigned short* __restrict__ out, int n) {
    int i = (blockIdx.x * 256 + threadIdx.x) * 4;
    if (i + 3 < n) {
        float4 f = *(const float4*)(in + i);
        union { unsigned short s[4]; uint2 u; } o;
        o.s[0] = f2bf(f.x); o.s[1] = f2bf(f.y); o.s[2] = f2bf(f.z); o.s[3] = f2bf(f.w);
        *(uint2*)(out + i) = o.u;
    }
}

// W [K x N] f32 row-major  ->  Wt [N x K] bf16 row-major
__global__ __launch_bounds__(256) void k_convT(const float* __restrict__ W,
                                               unsigned short* __restrict__ Wt, int K, int N) {
    int i = blockIdx.x * 256 + threadIdx.x;
    if (i < K * N) {
        int k = i / N, n = i - k * N;
        Wt[(size_t)n * K + k] = f2bf(W[i]);
    }
}

// ---------------- GEMM: C[M,N] = A[M,K](bf16) x Bt[N,K]^T(bf16) + bias, fused epilogues ----------------
// EPI 0: QKV scatter (bf16 Q[bh,s,hd], K[bh,s,hd], V^T[bh,hd,s])
// EPI 1: + resid (f32) -> f32 out
// EPI 2: ReLU -> bf16 out
// EPI 3: + resid (f32) -> f32 out
template <int EPI>
__global__ __launch_bounds__(256) void k_gemm(
    const unsigned short* __restrict__ A,
    const unsigned short* __restrict__ Bt,
    const float* __restrict__ bias,
    const float* __restrict__ resid,
    float* __restrict__ outF,
    unsigned short* __restrict__ out16,
    unsigned short* __restrict__ Qb,
    unsigned short* __restrict__ Kb,
    unsigned short* __restrict__ Vt,
    int Kd, int Nd)
{
    __shared__ unsigned short sA[2][128 * 40];
    __shared__ unsigned short sB[2][128 * 40];

    const int tid  = threadIdx.x;
    const int lane = tid & 31, wid = tid >> 5;
    const int rowBase = blockIdx.y * 128;
    const int colBase = blockIdx.x * 128;
    const int wRow = (wid & 3) * 32;
    const int wCol = (wid >> 2) * 64;
    const int hl  = lane >> 4;     // half-wave select
    const int l15 = lane & 15;

    v8f acc[2][4];
    #pragma unroll
    for (int i = 0; i < 2; ++i)
        #pragma unroll
        for (int j = 0; j < 4; ++j) acc[i][j] = vzero8();

    const int nkt = Kd / 32;

#if HAVE_TDM
    const unsigned short* Ab = A + (size_t)rowBase * Kd;
    const unsigned short* Bb = Bt + (size_t)colBase * Kd;
    const unsigned offA0 = lds_off(&sA[0][0]), offA1 = lds_off(&sA[1][0]);
    const unsigned offB0 = lds_off(&sB[0][0]), offB1 = lds_off(&sB[1][0]);
    if (wid == 0) {
        tdm_load_2d(offA0, Ab, 32u, 128u, (unsigned)Kd, 3u);  // pad every 16 DW -> stride 40 elems
        tdm_load_2d(offB0, Bb, 32u, 128u, (unsigned)Kd, 3u);
    }
#endif

    for (int kt = 0; kt < nkt; ++kt) {
        const int cur = kt & 1;
#if HAVE_TDM
        if (wid == 0) tdm_wait();        // DMA for buf[cur] complete
        __syncthreads();                 // all waves: data visible, prev buffer free
        if (wid == 0 && kt + 1 < nkt) {  // async prefetch of next K-slice into other buffer
            int k0n = (kt + 1) * 32;
            tdm_load_2d(cur ? offA0 : offA1, Ab + k0n, 32u, 128u, (unsigned)Kd, 3u);
            tdm_load_2d(cur ? offB0 : offB1, Bb + k0n, 32u, 128u, (unsigned)Kd, 3u);
        }
#else
        const int k0 = kt * 32;
        __syncthreads();
        for (int c = tid; c < 512; c += 256) {
            int r = c >> 2, kc = c & 3;
            const unsigned short* pa = A + (size_t)(rowBase + r) * Kd + k0 + kc * 8;
            const unsigned short* pb = Bt + (size_t)(colBase + r) * Kd + k0 + kc * 8;
            *(v8u16*)&sA[cur][r * 40 + kc * 8] = *(const v8u16*)pa;
            *(v8u16*)&sB[cur][r * 40 + kc * 8] = *(const v8u16*)pb;
            if (k0 + 32 < Kd) {
                __builtin_prefetch(pa + 32, 0, 0);
                __builtin_prefetch(pb + 32, 0, 0);
            }
        }
        __syncthreads();
#endif

        v16bf af[2], bfr[4];
        #pragma unroll
        for (int mi = 0; mi < 2; ++mi) {
            int r = wRow + mi * 16 + l15;
            int c0 = hl * 8;
            af[mi] = mkfrag(&sA[cur][r * 40 + c0], &sA[cur][r * 40 + c0 + 16]);
        }
        #pragma unroll
        for (int ni = 0; ni < 4; ++ni) {
            int n = wCol + ni * 16 + l15;
            int c0 = hl * 8;
            bfr[ni] = mkfrag(&sB[cur][n * 40 + c0], &sB[cur][n * 40 + c0 + 16]);
        }
        #pragma unroll
        for (int mi = 0; mi < 2; ++mi)
            #pragma unroll
            for (int ni = 0; ni < 4; ++ni)
                acc[mi][ni] = wmma_bf16(af[mi], bfr[ni], acc[mi][ni]);
    }

    #pragma unroll
    for (int mi = 0; mi < 2; ++mi) {
        #pragma unroll
        for (int ni = 0; ni < 4; ++ni) {
            int col = colBase + wCol + ni * 16 + l15;
            float bv = bias[col];
            #pragma unroll
            for (int r = 0; r < 8; ++r) {
                int row = rowBase + wRow + mi * 16 + r + hl * 8;
                float v = acc[mi][ni][r] + bv;
                if constexpr (EPI == 0) {
                    int b = row >> 10, s = row & 1023;
                    int h = col / 192, j = col - h * 192;
                    int bh = b * NH + h;
                    if (j < 64)
                        Qb[((size_t)(bh * Sq + s)) * HD + j] = f2bf(v);
                    else if (j < 128)
                        Kb[((size_t)(bh * Sq + s)) * HD + (j - 64)] = f2bf(v);
                    else
                        Vt[((size_t)(bh * HD + (j - 128))) * Sq + s] = f2bf(v);
                } else if constexpr (EPI == 1) {
                    size_t idx = (size_t)row * Nd + col;
                    outF[idx] = v + resid[idx];
                } else if constexpr (EPI == 2) {
                    out16[(size_t)row * Nd + col] = f2bf(v > 0.f ? v : 0.f);
                } else {
                    size_t idx = (size_t)row * Nd + col;
                    outF[idx] = v + resid[idx];
                }
            }
        }
    }
}

// ---------------- fused attention (flash-style, online softmax) ----------------
// grid: (S/128, B*H); block 256 (8 waves, each owns 16 query rows)
__global__ __launch_bounds__(256) void k_attn(
    const unsigned short* __restrict__ Q,   // [bh, s, hd] bf16
    const unsigned short* __restrict__ K,   // [bh, s, hd] bf16
    const unsigned short* __restrict__ Vt,  // [bh, hd, s] bf16
    const float* __restrict__ mask,         // [b, s, s] f32
    unsigned short* __restrict__ ctx)       // [b, s, d] bf16
{
    __shared__ unsigned short sK[128 * 72];   // [key][hd]
    __shared__ unsigned short sV[64 * 136];   // [hd][key]
    __shared__ unsigned short sP[8][16 * 72]; // per-wave, [qrow][key(64)]

    const int tid = threadIdx.x, lane = tid & 31, wid = tid >> 5;
    const int bh = blockIdx.y;
    const int b  = bh >> 4, h = bh & 15;
    const int qBase = blockIdx.x * 128;
    const int hl = lane >> 4, l15 = lane & 15;
    const int qRow = qBase + wid * 16;

    // Q fragments straight from global (row-major, b128 loads)
    const unsigned short* qp = Q + ((size_t)(bh * Sq + qRow + l15)) * HD;
    v16bf qa[2];
    #pragma unroll
    for (int f = 0; f < 2; ++f) {
        int c0 = f * 32 + hl * 8;
        qa[f] = mkfrag(qp + c0, qp + c0 + 16);
    }

    v8f oacc[4];
    #pragma unroll
    for (int i = 0; i < 4; ++i) oacc[i] = vzero8();
    float mrow[8], lrow[8];
    #pragma unroll
    for (int r = 0; r < 8; ++r) { mrow[r] = -3.0e38f; lrow[r] = 0.f; }

    unsigned short* sPw = sP[wid];

    for (int kt = 0; kt < 8; ++kt) {
        int kBase = kt * 128;
        __syncthreads();
#if HAVE_TDM
        if (wid == 0) {
            tdm_load_2d(lds_off(sK), K + ((size_t)(bh * Sq + kBase)) * HD,
                        64u, 128u, 64u, 4u);            // pad every 32 DW -> stride 72 elems
            tdm_load_2d(lds_off(sV), Vt + (size_t)bh * HD * Sq + kBase,
                        128u, 64u, (unsigned)Sq, 5u);   // pad every 64 DW -> stride 136 elems
            tdm_wait();
        }
        __syncthreads();
#else
        for (int c = tid; c < 1024; c += 256) {
            { int r = c >> 3, kc = c & 7;
              *(v8u16*)&sK[r * 72 + kc * 8] =
                  *(const v8u16*)(K + ((size_t)(bh * Sq + kBase + r)) * HD + kc * 8); }
            { int r = c >> 4, kc = c & 15;
              *(v8u16*)&sV[r * 136 + kc * 8] =
                  *(const v8u16*)(Vt + ((size_t)(bh * HD + r)) * Sq + kBase + kc * 8); }
        }
        __syncthreads();
#endif

        // scores: S = Q (16x64) x K^T (64x128)
        v8f sc[8];
        #pragma unroll
        for (int ni = 0; ni < 8; ++ni) {
            sc[ni] = vzero8();
            #pragma unroll
            for (int f = 0; f < 2; ++f) {
                int n  = ni * 16 + l15;
                int h0 = f * 32 + hl * 8;
                v16bf bk = mkfrag(&sK[n * 72 + h0], &sK[n * 72 + h0 + 16]);
                sc[ni] = wmma_bf16(qa[f], bk, sc[ni]);
            }
        }

        // scale + mask
        const float* mp = mask + (((size_t)b * Sq + qRow) * Sq + kBase);
        #pragma unroll
        for (int ni = 0; ni < 8; ++ni)
            #pragma unroll
            for (int r = 0; r < 8; ++r)
                sc[ni][r] = sc[ni][r] * 0.125f + mp[(size_t)(r + hl * 8) * Sq + ni * 16 + l15];

        // online softmax row stats (rows split across lane halves)
        #pragma unroll
        for (int r = 0; r < 8; ++r) {
            float mx = sc[0][r];
            #pragma unroll
            for (int ni = 1; ni < 8; ++ni) mx = fmaxf(mx, sc[ni][r]);
            mx = fmaxf(mx, __shfl_xor(mx, 1, 32));
            mx = fmaxf(mx, __shfl_xor(mx, 2, 32));
            mx = fmaxf(mx, __shfl_xor(mx, 4, 32));
            mx = fmaxf(mx, __shfl_xor(mx, 8, 32));
            float mnew  = fmaxf(mrow[r], mx);
            float alpha = exp2f((mrow[r] - mnew) * 1.44269504f);
            mrow[r] = mnew;
            float ps = 0.f;
            #pragma unroll
            for (int ni = 0; ni < 8; ++ni) {
                float p = exp2f((sc[ni][r] - mnew) * 1.44269504f);
                sc[ni][r] = p;
                ps += p;
            }
            ps += __shfl_xor(ps, 1, 32);
            ps += __shfl_xor(ps, 2, 32);
            ps += __shfl_xor(ps, 4, 32);
            ps += __shfl_xor(ps, 8, 32);
            lrow[r] = lrow[r] * alpha + ps;
            #pragma unroll
            for (int n2 = 0; n2 < 4; ++n2) oacc[n2][r] *= alpha;
        }

        // P @ V in two 64-key halves (C-layout -> A-layout via per-wave LDS)
        #pragma unroll
        for (int kh = 0; kh < 2; ++kh) {
            #pragma unroll
            for (int ni = 0; ni < 4; ++ni) {
                int nig = kh * 4 + ni;
                #pragma unroll
                for (int r = 0; r < 8; ++r)
                    sPw[(r + hl * 8) * 72 + ni * 16 + l15] = f2bf(sc[nig][r]);
            }
            #pragma unroll
            for (int kk = 0; kk < 2; ++kk) {
                int c0 = kk * 32 + hl * 8;
                v16bf ap = mkfrag(&sPw[l15 * 72 + c0], &sPw[l15 * 72 + c0 + 16]);
                #pragma unroll
                for (int n2 = 0; n2 < 4; ++n2) {
                    int hr  = n2 * 16 + l15;
                    int kv0 = kh * 64 + kk * 32 + hl * 8;
                    v16bf bv = mkfrag(&sV[hr * 136 + kv0], &sV[hr * 136 + kv0 + 16]);
                    oacc[n2] = wmma_bf16(ap, bv, oacc[n2]);
                }
            }
        }
    }

    // finalize: O /= l, store bf16 context [b, s, h*64 + d]
    #pragma unroll
    for (int r = 0; r < 8; ++r) {
        float inv = 1.f / lrow[r];
        int row = qRow + r + hl * 8;
        #pragma unroll
        for (int n2 = 0; n2 < 4; ++n2) {
            float v = oacc[n2][r] * inv;
            ctx[((size_t)(b * Sq + row)) * Dm + h * 64 + n2 * 16 + l15] = f2bf(v);
        }
    }
}

// ---------------- layer norm (one row per block) ----------------
__global__ __launch_bounds__(256) void k_ln(const float* __restrict__ in,
                                            const float* __restrict__ gamma,
                                            const float* __restrict__ beta,
                                            float* __restrict__ outF,
                                            unsigned short* __restrict__ outBf) {
    __shared__ float s1[256], s2[256];
    const int row = blockIdx.x, tid = threadIdx.x;
    const float* p = in + (size_t)row * Dm;
    float4 v = *(const float4*)(p + tid * 4);
    s1[tid] = v.x + v.y + v.z + v.w;
    s2[tid] = v.x * v.x + v.y * v.y + v.z * v.z + v.w * v.w;
    __syncthreads();
    for (int st = 128; st > 0; st >>= 1) {
        if (tid < st) { s1[tid] += s1[tid + st]; s2[tid] += s2[tid + st]; }
        __syncthreads();
    }
    float mean = s1[0] * (1.f / Dm);
    float var  = s2[0] * (1.f / Dm) - mean * mean;
    float rstd = rsqrtf(var + 1e-5f);
    float4 g  = *(const float4*)(gamma + tid * 4);
    float4 be = *(const float4*)(beta + tid * 4);
    float4 o;
    o.x = g.x * (v.x - mean) * rstd + be.x;
    o.y = g.y * (v.y - mean) * rstd + be.y;
    o.z = g.z * (v.z - mean) * rstd + be.z;
    o.w = g.w * (v.w - mean) * rstd + be.w;
    *(float4*)(outF + (size_t)row * Dm + tid * 4) = o;
    if (outBf) {
        union { unsigned short s[4]; uint2 u; } ob;
        ob.s[0] = f2bf(o.x); ob.s[1] = f2bf(o.y); ob.s[2] = f2bf(o.z); ob.s[3] = f2bf(o.w);
        *(uint2*)(outBf + (size_t)row * Dm + tid * 4) = ob.u;
    }
}

// ---------------- host-side orchestration ----------------
extern "C" void kernel_launch(void* const* d_in, const int* in_sizes, int n_in,
                              void* d_out, int out_size, void* d_ws, size_t ws_size,
                              hipStream_t stream) {
    const float* x     = (const float*)d_in[0];
    const float* mask  = (const float*)d_in[1];
    const float* w_qkv = (const float*)d_in[2];
    const float* b_qkv = (const float*)d_in[3];
    const float* w_out = (const float*)d_in[4];
    const float* b_out = (const float*)d_in[5];
    const float* g1    = (const float*)d_in[6];
    const float* be1   = (const float*)d_in[7];
    const float* w1    = (const float*)d_in[8];
    const float* b1    = (const float*)d_in[9];
    const float* w2    = (const float*)d_in[10];
    const float* b2    = (const float*)d_in[11];
    const float* g2    = (const float*)d_in[12];
    const float* be2   = (const float*)d_in[13];
    float* out = (float*)d_out;
    char*  ws  = (char*)d_ws;

    constexpr size_t MB = 1024u * 1024u;
    unsigned short* Xbf   = (unsigned short*)(ws + 0);
    unsigned short* WqkvT = (unsigned short*)(ws + 8 * MB);
    unsigned short* WoutT = (unsigned short*)(ws + 14 * MB);
    unsigned short* W1T   = (unsigned short*)(ws + 16 * MB);
    unsigned short* W2T   = (unsigned short*)(ws + 24 * MB);
    unsigned short* Qb    = (unsigned short*)(ws + 32 * MB);
    unsigned short* Kb    = (unsigned short*)(ws + 40 * MB);
    unsigned short* Vt    = (unsigned short*)(ws + 48 * MB);
    unsigned short* Ctx   = (unsigned short*)(ws + 56 * MB);
    unsigned short* Hff   = (unsigned short*)(ws + 32 * MB); // reuses Q/K/V/Ctx after attention
    float*          hbuf  = (float*)(ws + 64 * MB);
    float*          x1f   = (float*)(ws + 80 * MB);
    unsigned short* x1bf  = (unsigned short*)(ws + 0);       // reuses Xbf slot

    dim3 blk(256);

    // casts
    k_f32_to_bf16<<<(Mrows * Dm / 4 + 255) / 256, blk, 0, stream>>>(x, Xbf, Mrows * Dm);
    k_convT<<<(1024 * 3072 + 255) / 256, blk, 0, stream>>>(w_qkv, WqkvT, 1024, 3072);
    k_convT<<<(1024 * 1024 + 255) / 256, blk, 0, stream>>>(w_out, WoutT, 1024, 1024);
    k_convT<<<(1024 * 4096 + 255) / 256, blk, 0, stream>>>(w1, W1T, 1024, 4096);
    k_convT<<<(4096 * 1024 + 255) / 256, blk, 0, stream>>>(w2, W2T, 4096, 1024);

    // QKV projection with fused scatter (V stored transposed)
    k_gemm<0><<<dim3(3072 / 128, Mrows / 128), blk, 0, stream>>>(
        Xbf, WqkvT, b_qkv, nullptr, nullptr, nullptr, Qb, Kb, Vt, 1024, 3072);

    // fused attention
    k_attn<<<dim3(Sq / 128, Bsz * NH), blk, 0, stream>>>(Qb, Kb, Vt, mask, Ctx);

    // output projection + residual (f32)
    k_gemm<1><<<dim3(1024 / 128, Mrows / 128), blk, 0, stream>>>(
        Ctx, WoutT, b_out, x, hbuf, nullptr, nullptr, nullptr, nullptr, 1024, 1024);

    // LN1 -> f32 + bf16 copies
    k_ln<<<Mrows, blk, 0, stream>>>(hbuf, g1, be1, x1f, x1bf);

    // FFN1 + ReLU -> bf16
    k_gemm<2><<<dim3(4096 / 128, Mrows / 128), blk, 0, stream>>>(
        x1bf, W1T, b1, nullptr, nullptr, Hff, nullptr, nullptr, nullptr, 1024, 4096);

    // FFN2 + residual (f32)
    k_gemm<3><<<dim3(1024 / 128, Mrows / 128), blk, 0, stream>>>(
        Hff, W2T, b2, x1f, hbuf, nullptr, nullptr, nullptr, nullptr, 4096, 1024);

    // LN2 -> output
    k_ln<<<Mrows, blk, 0, stream>>>(hbuf, g2, be2, out, nullptr);
}
